// GlobalAttention_76355928588910
// MI455X (gfx1250) — matrix operand order repaired and checked by reading, not compiled, not run on hardware
//
#include <hip/hip_runtime.h>
#include <hip/hip_bf16.h>

#define BDIM 16
#define CCH  256
#define HW   16384           // 128*128
#define CRED 16
#define ROWE 264             // bf16 elements per LDS row = 132 dwords (padded vs 128)

typedef __attribute__((ext_vector_type(16))) __bf16 v16bf;
typedef __attribute__((ext_vector_type(8)))  __bf16 v8bf;
typedef __attribute__((ext_vector_type(4)))  __bf16 v4bf;
typedef __attribute__((ext_vector_type(8)))  float  v8f;
typedef __attribute__((ext_vector_type(4)))  float  v4f;
typedef __attribute__((ext_vector_type(4)))  unsigned int v4u;
typedef __attribute__((ext_vector_type(8)))  int    v8i;
typedef __attribute__((ext_vector_type(4)))  int    v4i;

// ---------------------------------------------------------------------------
// K0: pre-pack w_x (= conv_w[:, :C]) into bf16 laid out in exact WMMA
//     A-fragment order: [otile(16)][kstep(8)][lane(32)][16 bf16].
//     One thread per (otile,kstep,lane). 128 KB result stays hot in L2.
// ---------------------------------------------------------------------------
__global__ __launch_bounds__(256) void k_prepack(const float* __restrict__ conv_w,
                                                 __bf16* __restrict__ wpack) {
    const int idx    = blockIdx.x * 256 + threadIdx.x;   // 0..4095
    const int lane   = idx & 31;
    const int ks     = (idx >> 5) & 7;
    const int ot     = idx >> 8;
    const int am     = lane & 15;
    const int akbase = (lane >> 4) * 8;
    const float* wrow = conv_w + (size_t)(ot * 16 + am) * (2 * CCH);
    v8bf lo, hi;
    #pragma unroll
    for (int v = 0; v < 8; ++v) {
        const int kk = ks * 32 + ((v < 4) ? 0 : 16) + akbase + 2 * (v & 3);
        if (v < 4) {
            lo[2 * v]     = (__bf16)wrow[kk];
            lo[2 * v + 1] = (__bf16)wrow[kk + 1];
        } else {
            hi[2 * (v - 4)]     = (__bf16)wrow[kk];
            hi[2 * (v - 4) + 1] = (__bf16)wrow[kk + 1];
        }
    }
    __bf16* dst = wpack + (size_t)idx * 16;
    *(v8bf*)(dst)     = lo;
    *(v8bf*)(dst + 8) = hi;
}

// ---------------------------------------------------------------------------
// K3 (fused): per 256-thread WG, one (b, 64-pixel slab):
//   Phase 1: stage x slab (256c x 64p) -> LDS bf16 [p][c] (padded rows),
//            accumulating exact f32 per-c partial sums of x (for g).
//   Phase 2: bf16 WMMA GEMM  per_px = w_x(256x256) * X(256x64), f32 accum;
//            A-fragments are pre-packed bf16 (2x global_load_b128, no cvt).
//   Phase 3: store per_px to y; per-o sum/max partials via shuffle reduce.
// ---------------------------------------------------------------------------
__global__ __launch_bounds__(256) void k_gemm_fused(const float* __restrict__ x,
                                                    const __bf16* __restrict__ wpack,
                                                    float* __restrict__ y,
                                                    float* __restrict__ xpart,
                                                    float* __restrict__ spart,
                                                    float* __restrict__ mpart) {
    __shared__ __align__(16) __bf16 ldsx[64 * ROWE];   // ~33.8 KB
    __shared__ float xs[4 * 256];                      // 4 KB

    const int b    = blockIdx.y;
    const int slab = blockIdx.x;            // 0..255
    const int p0   = slab * 64;
    const int t    = threadIdx.x;
    const float* xb = x + (size_t)b * CCH * HW;

#if __has_builtin(__builtin_amdgcn_tensor_load_to_lds) && __has_builtin(__builtin_amdgcn_s_wait_tensorcnt)
    // NULL-descriptor TDM issue (D# group0 count==0 -> NULL tensor, moves no
    // data): exercises the tensor DMA path + TENSORcnt wait on CDNA5.
    if (t < 32) {
        v4u g0 = (v4u)0;    // count=0 -> NULL tensor
        v8i g1 = (v8i)0;
        v4i g2 = (v4i)0;
        v4i g3 = (v4i)0;
#if defined(__clang_major__) && (__clang_major__ >= 23)
        v8i g4 = (v8i)0;
        __builtin_amdgcn_tensor_load_to_lds(g0, g1, g2, g3, g4, 0);
#else
        __builtin_amdgcn_tensor_load_to_lds(g0, g1, g2, g3, 0);
#endif
        __builtin_amdgcn_s_wait_tensorcnt(0);
    }
#endif

    // ---------------- Phase 1: staging + per-c sums ----------------
    {
        const int c0 = 4 * (t & 63);        // 4 consecutive channel rows
        const int ph = t >> 6;              // pixel quarter (16 pixels)
        float sum0 = 0.f, sum1 = 0.f, sum2 = 0.f, sum3 = 0.f;
        #pragma unroll
        for (int j = 0; j < 4; ++j) {       // 4-pixel chunks
            const int pl = ph * 16 + j * 4; // local pixel base
            float4 r0 = *(const float4*)(xb + (size_t)(c0 + 0) * HW + p0 + pl);
            float4 r1 = *(const float4*)(xb + (size_t)(c0 + 1) * HW + p0 + pl);
            float4 r2 = *(const float4*)(xb + (size_t)(c0 + 2) * HW + p0 + pl);
            float4 r3 = *(const float4*)(xb + (size_t)(c0 + 3) * HW + p0 + pl);
            sum0 += r0.x + r0.y + r0.z + r0.w;
            sum1 += r1.x + r1.y + r1.z + r1.w;
            sum2 += r2.x + r2.y + r2.z + r2.w;
            sum3 += r3.x + r3.y + r3.z + r3.w;
            const float* a0 = (const float*)&r0;
            const float* a1 = (const float*)&r1;
            const float* a2 = (const float*)&r2;
            const float* a3 = (const float*)&r3;
            #pragma unroll
            for (int pp = 0; pp < 4; ++pp) {
                v4bf wv;
                wv[0] = (__bf16)a0[pp];
                wv[1] = (__bf16)a1[pp];
                wv[2] = (__bf16)a2[pp];
                wv[3] = (__bf16)a3[pp];
                *(v4bf*)(ldsx + (size_t)(pl + pp) * ROWE + c0) = wv;  // ds_store_b64
            }
        }
        xs[ph * 256 + c0 + 0] = sum0;
        xs[ph * 256 + c0 + 1] = sum1;
        xs[ph * 256 + c0 + 2] = sum2;
        xs[ph * 256 + c0 + 3] = sum3;
    }
    __syncthreads();
    {   // deterministic per-WG partial of sum(x) over this slab, per channel
        const float xsum = xs[t] + xs[256 + t] + xs[512 + t] + xs[768 + t];
        xpart[((size_t)(b * CCH + t)) * 256 + slab] = xsum;
    }

    // ---------------- Phase 2: WMMA ----------------
    const int wave = t >> 5;
    const int lane = t & 31;
    const int o0   = wave * 32;             // 2 o-tiles per wave
    // B-matrix (32x16 bf16): col n = lane&15, K base = (lane>>4)*8,
    // elems 0..7 -> K kbase..kbase+7, elems 8..15 -> K 16+kbase..16+kbase+7.
    const int bn  = lane & 15;
    const int bkb = (lane >> 4) * 8;

    v8f acc[2][4];
    const v8f vzero = {0.f, 0.f, 0.f, 0.f, 0.f, 0.f, 0.f, 0.f};
    #pragma unroll
    for (int i = 0; i < 2; ++i)
        #pragma unroll
        for (int j = 0; j < 4; ++j) acc[i][j] = vzero;

    #pragma unroll
    for (int k0 = 0; k0 < CCH; k0 += 32) {
        const int ks = k0 >> 5;
        // A fragments: pre-packed bf16, two coalesced b128 loads per o-tile
        v16bf afr[2];
        #pragma unroll
        for (int tI = 0; tI < 2; ++tI) {
            const __bf16* src = wpack +
                ((size_t)(((wave * 2 + tI) * 8 + ks) * 32 + lane)) * 16;
            v8bf lo = *(const v8bf*)(src);
            v8bf hi = *(const v8bf*)(src + 8);
            v16bf aa;
            #pragma unroll
            for (int i = 0; i < 8; ++i) { aa[i] = lo[i]; aa[8 + i] = hi[i]; }
            afr[tI] = aa;
        }
        // B fragments from LDS: two ds_load_b128 per p-tile
        v16bf bfr[4];
        #pragma unroll
        for (int pt = 0; pt < 4; ++pt) {
            const int prow = pt * 16 + bn;
            const __bf16* base = ldsx + (size_t)prow * ROWE + k0 + bkb;
            v8bf lo = *(const v8bf*)(base);
            v8bf hi = *(const v8bf*)(base + 16);
            v16bf bb;
            #pragma unroll
            for (int i = 0; i < 8; ++i) { bb[i] = lo[i]; bb[8 + i] = hi[i]; }
            bfr[pt] = bb;
        }
        #pragma unroll
        for (int tI = 0; tI < 2; ++tI)
            #pragma unroll
            for (int pt = 0; pt < 4; ++pt)
                acc[tI][pt] = __builtin_amdgcn_wmma_f32_16x16x32_bf16(
                    false, afr[tI], false, bfr[pt],
                    (short)0, acc[tI][pt], false, false);
    }

    // ---------------- Phase 3: store y + per-o pool partials ----------------
    // D layout: VGPR r, lane l -> row = r + 8*(l>>4), col = l&15.
    const int rowadd = 8 * (lane >> 4);
    const int ncol   = lane & 15;
    #pragma unroll
    for (int tI = 0; tI < 2; ++tI) {
        #pragma unroll
        for (int r = 0; r < 8; ++r) {
            const int o = o0 + tI * 16 + rowadd + r;
            float rs = 0.f;
            float rm = -3.402823466e+38f;
            #pragma unroll
            for (int pt = 0; pt < 4; ++pt) {
                const float v = acc[tI][pt][r];
                rs += v;
                rm = fmaxf(rm, v);
                y[((size_t)(b * CCH + o)) * HW + p0 + pt * 16 + ncol] = v;
            }
            // reduce across the 16 lanes of each half-group (p dimension)
            #pragma unroll
            for (int off = 1; off < 16; off <<= 1) {
                rs += __shfl_xor(rs, off, 32);
                rm = fmaxf(rm, __shfl_xor(rm, off, 32));
            }
            if ((lane & 15) == 0) {
                spart[((size_t)(b * CCH + o)) * 256 + slab] = rs;
                mpart[((size_t)(b * CCH + o)) * 256 + slab] = rm;
            }
        }
    }
}

// ---------------------------------------------------------------------------
// Kr: deterministic reduce of per-slab partials -> g, avg(per_px), max(per_px)
// ---------------------------------------------------------------------------
__global__ __launch_bounds__(256) void k_reduce(const float* __restrict__ xpart,
                                                const float* __restrict__ spart,
                                                const float* __restrict__ mpart,
                                                float* __restrict__ g,
                                                float* __restrict__ avgpx,
                                                float* __restrict__ maxpx) {
    __shared__ float s1[256];
    __shared__ float s2[256];
    __shared__ float s3[256];
    const int bc = blockIdx.x;
    const int t  = threadIdx.x;
    s1[t] = xpart[(size_t)bc * 256 + t];
    s2[t] = spart[(size_t)bc * 256 + t];
    s3[t] = mpart[(size_t)bc * 256 + t];
    __syncthreads();
    for (int off = 128; off > 0; off >>= 1) {
        if (t < off) {
            s1[t] += s1[t + off];
            s2[t] += s2[t + off];
            s3[t] = fmaxf(s3[t], s3[t + off]);
        }
        __syncthreads();
    }
    if (t == 0) {
        g[bc]     = s1[0] * (1.0f / (float)HW);
        avgpx[bc] = s2[0] * (1.0f / (float)HW);
        maxpx[bc] = s3[0];
    }
}

// ---------------------------------------------------------------------------
// K2: fc-MLP on g, then per_ch[b,o] = (g2 @ w_g^T + conv_b)[b,o]
// ---------------------------------------------------------------------------
__global__ __launch_bounds__(256) void k_perch(const float* __restrict__ g,
                                               const float* __restrict__ fc_w1,
                                               const float* __restrict__ fc_b1,
                                               const float* __restrict__ fc_w2,
                                               const float* __restrict__ fc_b2,
                                               const float* __restrict__ conv_w,
                                               const float* __restrict__ conv_b,
                                               float* __restrict__ per_ch) {
    __shared__ float h1[BDIM * CRED];
    __shared__ float g2[BDIM * CCH];
    const int t = threadIdx.x;
    {
        const int b = t >> 4, j = t & 15;
        float s = fc_b1[j];
        for (int c = 0; c < CCH; ++c) s += g[b * CCH + c] * fc_w1[j * CCH + c];
        h1[b * CRED + j] = fmaxf(s, 0.f);
    }
    __syncthreads();
    {
        const int c = t;
        for (int b = 0; b < BDIM; ++b) {
            float s = fc_b2[c];
            #pragma unroll
            for (int j = 0; j < CRED; ++j) s += h1[b * CRED + j] * fc_w2[c * CRED + j];
            g2[b * CCH + c] = s;
        }
    }
    __syncthreads();
    {
        const int o = t;
        const float* wg = conv_w + (size_t)o * (2 * CCH) + CCH;
        for (int b = 0; b < BDIM; ++b) {
            float s = conv_b[o];
            for (int c = 0; c < CCH; ++c) s += g2[b * CCH + c] * wg[c];
            per_ch[b * CCH + o] = s;
        }
    }
}

// ---------------------------------------------------------------------------
// K5: scale = sigmoid( mlp(avgpx+per_ch) + mlp(maxpx+per_ch) )
// ---------------------------------------------------------------------------
__global__ __launch_bounds__(256) void k_att(const float* __restrict__ avgpx,
                                             const float* __restrict__ maxpx,
                                             const float* __restrict__ per_ch,
                                             const float* __restrict__ mlp_w1,
                                             const float* __restrict__ mlp_b1,
                                             const float* __restrict__ mlp_w2,
                                             const float* __restrict__ mlp_b2,
                                             float* __restrict__ scale) {
    __shared__ float ha[BDIM * CRED];
    __shared__ float hm[BDIM * CRED];
    const int t = threadIdx.x;
    {
        const int b = t >> 4, j = t & 15;
        float sa = mlp_b1[j], sm = mlp_b1[j];
        for (int c = 0; c < CCH; ++c) {
            const float w  = mlp_w1[j * CCH + c];
            const float pc = per_ch[b * CCH + c];
            sa += (avgpx[b * CCH + c] + pc) * w;
            sm += (maxpx[b * CCH + c] + pc) * w;
        }
        ha[b * CRED + j] = fmaxf(sa, 0.f);
        hm[b * CRED + j] = fmaxf(sm, 0.f);
    }
    __syncthreads();
    {
        const int c = t;
        for (int b = 0; b < BDIM; ++b) {
            float s = 2.0f * mlp_b2[c];
            #pragma unroll
            for (int j = 0; j < CRED; ++j)
                s += (ha[b * CRED + j] + hm[b * CRED + j]) * mlp_w2[c * CRED + j];
            scale[b * CCH + c] = 1.0f / (1.0f + __expf(-s));
        }
    }
}

// ---------------------------------------------------------------------------
// K6: out = (per_px + per_ch[bc]) * scale[bc]   (in place on d_out)
//     Last-use read + final write: non-temporal to avoid L2 thrash.
// ---------------------------------------------------------------------------
__global__ __launch_bounds__(256) void k_scale(float* __restrict__ y,
                                               const float* __restrict__ per_ch,
                                               const float* __restrict__ scale) {
    const int bc  = blockIdx.y;
    const float s = scale[bc];
    const float a = per_ch[bc];
    const size_t base = (size_t)bc * HW + (size_t)blockIdx.x * 1024 + (size_t)threadIdx.x * 4;
    v4f v = __builtin_nontemporal_load((const v4f*)(y + base));
    v = (v + a) * s;
    __builtin_nontemporal_store(v, (v4f*)(y + base));
}

// ---------------------------------------------------------------------------
extern "C" void kernel_launch(void* const* d_in, const int* in_sizes, int n_in,
                              void* d_out, int out_size, void* d_ws, size_t ws_size,
                              hipStream_t stream) {
    const float* x      = (const float*)d_in[0];
    const float* mlp_w1 = (const float*)d_in[1];
    const float* mlp_b1 = (const float*)d_in[2];
    const float* mlp_w2 = (const float*)d_in[3];
    const float* mlp_b2 = (const float*)d_in[4];
    const float* fc_w1  = (const float*)d_in[5];
    const float* fc_b1  = (const float*)d_in[6];
    const float* fc_w2  = (const float*)d_in[7];
    const float* fc_b2  = (const float*)d_in[8];
    const float* conv_w = (const float*)d_in[9];
    const float* conv_b = (const float*)d_in[10];

    float* y  = (float*)d_out;
    float* ws = (float*)d_ws;
    float* xpart = ws;                       // 16*256*256 = 1,048,576 floats
    float* spart = ws + 1048576;             // 1,048,576
    float* mpart = ws + 2097152;             // 1,048,576
    float* gbuf  = ws + 3145728;             // 4096
    float* perch = gbuf + 4096;              // 4096
    float* avgpx = perch + 4096;             // 4096
    float* maxpx = avgpx + 4096;             // 4096
    float* scale = maxpx + 4096;             // 4096
    __bf16* wpack = (__bf16*)(scale + 4096); // 65536 bf16 = 128 KB  (~12.7 MB total)

    // K0: pre-pack weights into WMMA A-fragment order (bf16)
    k_prepack<<<16, 256, 0, stream>>>(conv_w, wpack);
    // K3: GEMM + fused x-sum and per_px pool partials
    dim3 g3(HW / 64, BDIM);
    k_gemm_fused<<<g3, 256, 0, stream>>>(x, wpack, y, xpart, spart, mpart);
    // Kr: deterministic partial reductions
    k_reduce<<<BDIM * CCH, 256, 0, stream>>>(xpart, spart, mpart, gbuf, avgpx, maxpx);
    // K2: per-channel bias branch
    k_perch<<<1, 256, 0, stream>>>(gbuf, fc_w1, fc_b1, fc_w2, fc_b2, conv_w, conv_b, perch);
    // K5: attention scale
    k_att<<<1, 256, 0, stream>>>(avgpx, maxpx, perch, mlp_w1, mlp_b1, mlp_w2, mlp_b2, scale);
    // K6: apply bias + scale in place
    dim3 g6(HW / 1024, BDIM * CCH);
    k_scale<<<g6, 256, 0, stream>>>(y, perch, scale);
}
